// LKA3d_Deform_Channel_sequential_13417477833354
// MI455X (gfx1250) — matrix operand order repaired and checked by reading, not compiled, not run on hardware
//
#include <hip/hip_runtime.h>
#include <hip/hip_bf16.h>

typedef __attribute__((ext_vector_type(16))) __bf16 v16bf;
typedef __attribute__((ext_vector_type(8)))  float  v8f;
typedef unsigned short ushort_t;

#define V3    13824   // 24*24*24
#define DIMS  24
#define PLANE 576     // 24*24
#define KT_BIG 108    // 3456/32
#define KT_ONE 4      // 128/32

// round-to-nearest-even float -> bf16 bits
__device__ __forceinline__ ushort_t f2bfu(float f) {
    unsigned u = __builtin_bit_cast(unsigned, f);
    u += 0x7FFFu + ((u >> 16) & 1u);
    return (ushort_t)(u >> 16);
}

// ---------------------------------------------------------------------------
// Weight pre-swizzle into per-lane A-fragment layout (16-bit A 16x32):
//   M = lane&15 (+16*mtile),  K = (h&7) + 16*(h>>3) + 8*(lane>>4)  (+32*kt)
// ---------------------------------------------------------------------------
#define N_FDEF (8*KT_BIG*512)
#define N_FOFF (6*KT_BIG*512)
#define N_F1   (8*KT_ONE*512)

__global__ __launch_bounds__(256) void prep_frags(
    const float* __restrict__ wdef, const float* __restrict__ woff,
    const float* __restrict__ w1,
    ushort_t* __restrict__ fdef, ushort_t* __restrict__ foff,
    ushort_t* __restrict__ f1)
{
    int idx = blockIdx.x * 256 + threadIdx.x;
    if (idx < N_FDEF) {
        int mt = idx / (KT_BIG*512), rem = idx % (KT_BIG*512);
        int kt = rem / 512, q = rem % 512;
        int lane = q >> 4, h = q & 15;
        int m  = mt*16 + (lane & 15);
        int kk = (h & 7) + ((h >> 3) << 4) + ((lane >> 4) << 3);
        int k  = kt*32 + kk;
        int c = k / 27, tap = k % 27;
        fdef[idx] = f2bfu(wdef[(m*128 + c)*27 + tap]);
    }
    int i2 = idx - N_FDEF;
    if (i2 >= 0 && i2 < N_FOFF) {
        int mt = i2 / (KT_BIG*512), rem = i2 % (KT_BIG*512);
        int kt = rem / 512, q = rem % 512;
        int lane = q >> 4, h = q & 15;
        int m  = mt*16 + (lane & 15);
        int kk = (h & 7) + ((h >> 3) << 4) + ((lane >> 4) << 3);
        int k  = kt*32 + kk;
        int c = k / 27, tap = k % 27;
        float v = (m < 81) ? woff[(m*128 + c)*27 + tap] : 0.0f;
        foff[i2] = f2bfu(v);
    }
    int i3 = idx - (N_FDEF + N_FOFF);
    if (i3 >= 0 && i3 < N_F1) {
        int mt = i3 / (KT_ONE*512), rem = i3 % (KT_ONE*512);
        int kt = rem / 512, q = rem % 512;
        int lane = q >> 4, h = q & 15;
        int m  = mt*16 + (lane & 15);
        int kk = (h & 7) + ((h >> 3) << 4) + ((lane >> 4) << 3);
        int k  = kt*32 + kk;
        f1[i3] = f2bfu(w1[m*128 + k]);
    }
}

// ---------------------------------------------------------------------------
// Depthwise 5x5x5 conv, pad = 2*dil (size preserving). grid = (54, 128)
// ---------------------------------------------------------------------------
__global__ __launch_bounds__(256) void dw5(
    const float* __restrict__ in, const float* __restrict__ w,
    const float* __restrict__ b, float* __restrict__ out, int dil)
{
    const int c = blockIdx.y;
    const int v = blockIdx.x * 256 + threadIdx.x;
    const int z = v / PLANE, y = (v % PLANE) / DIMS, x = v % DIMS;
    const float* wc  = w  + c * 125;
    const float* inc = in + c * V3;
    float acc = b[c];
    for (int kd = 0; kd < 5; ++kd) {
        int iz = z + (kd - 2) * dil;
        if ((unsigned)iz >= (unsigned)DIMS) continue;
        for (int kh = 0; kh < 5; ++kh) {
            int iy = y + (kh - 2) * dil;
            if ((unsigned)iy >= (unsigned)DIMS) continue;
            for (int kw = 0; kw < 5; ++kw) {
                int ix = x + (kw - 2) * dil;
                if ((unsigned)ix >= (unsigned)DIMS) continue;
                acc += wc[(kd*5 + kh)*5 + kw] * inc[(iz*DIMS + iy)*DIMS + ix];
            }
        }
    }
    out[c * V3 + v] = acc;
}

// B-fragment mapping (16-bit B 32x16): N = lane&15, K = (h&7) + 16*(h>>3) + 8*(lane>>4)
__device__ __forceinline__ void b_elem(int e, int& nn, int& kk) {
    int lane = e >> 4, h = e & 15;
    nn = lane & 15;
    kk = (h & 7) + ((h >> 3) << 4) + ((lane >> 4) << 3);
}

// ---------------------------------------------------------------------------
// Offset conv: implicit GEMM  [96(81) x 3456] x [3456 x 64-col tile]
// grid = 216 blocks x 256 threads; waves 0..5 own M-tiles, 4 N-subtiles each
// ---------------------------------------------------------------------------
__global__ __launch_bounds__(256) void off_gemm(
    const float* __restrict__ attn2, const ushort_t* __restrict__ afrag,
    const float* __restrict__ bias, float* __restrict__ offs)
{
    __shared__ __align__(32) ushort_t bfrag[2][4][512];
    const int col0 = blockIdx.x * 64, tid = threadIdx.x;
    const int lane = tid & 31, wv = tid >> 5;

    // B tile builder for K-tile kt into buffer p
    auto build = [&](int p, int kt) {
        for (int e4 = tid; e4 < 4*512; e4 += 256) {
            int st = e4 >> 9, e = e4 & 511;
            int nn, kk; b_elem(e, nn, kk);
            int k = kt*32 + kk;
            int c = k / 27, tap = k % 27;
            int n = col0 + st*16 + nn;
            int z = n / PLANE, y = (n % PLANE) / DIMS, x = n % DIMS;
            int iz = z + tap/9 - 1, iy = y + (tap/3)%3 - 1, ix = x + tap%3 - 1;
            float v = 0.0f;
            if ((unsigned)iz < DIMS && (unsigned)iy < DIMS && (unsigned)ix < DIMS)
                v = attn2[c*V3 + (iz*DIMS + iy)*DIMS + ix];
            bfrag[p][st][e] = f2bfu(v);
        }
    };

    v8f acc0 = {}, acc1 = {}, acc2 = {}, acc3 = {};
    build(0, 0);
    __syncthreads();
    for (int kt = 0; kt < KT_BIG; ++kt) {
        int cur = kt & 1;
        if (wv < 6) {
            v16bf a = *(const v16bf*)(afrag + ((size_t)(wv*KT_BIG + kt)*512 + lane*16));
            v16bf b0 = *(const v16bf*)(&bfrag[cur][0][lane*16]);
            v16bf b1 = *(const v16bf*)(&bfrag[cur][1][lane*16]);
            v16bf b2 = *(const v16bf*)(&bfrag[cur][2][lane*16]);
            v16bf b3 = *(const v16bf*)(&bfrag[cur][3][lane*16]);
            acc0 = __builtin_amdgcn_wmma_f32_16x16x32_bf16(false, a, false, b0, (short)0, acc0, false, false);
            acc1 = __builtin_amdgcn_wmma_f32_16x16x32_bf16(false, a, false, b1, (short)0, acc1, false, false);
            acc2 = __builtin_amdgcn_wmma_f32_16x16x32_bf16(false, a, false, b2, (short)0, acc2, false, false);
            acc3 = __builtin_amdgcn_wmma_f32_16x16x32_bf16(false, a, false, b3, (short)0, acc3, false, false);
        }
        if (kt + 1 < KT_BIG) build(cur ^ 1, kt + 1);
        __syncthreads();
    }
    if (wv < 6) {
        int mhi = (lane >> 4) << 3;
        v8f accs[4] = {acc0, acc1, acc2, acc3};
        for (int st = 0; st < 4; ++st) {
            int n = col0 + st*16 + (lane & 15);
            for (int r = 0; r < 8; ++r) {
                int m = wv*16 + r + mhi;
                if (m < 81) offs[m*V3 + n] = accs[st][r] + bias[m];
            }
        }
    }
}

// ---------------------------------------------------------------------------
// Deformable 3x3x3: trilinear sample + implicit GEMM [128 x 3456] x [3456 x 32]
// grid = 432 x 256 (8 waves = 8 M-tiles, 2 N-subtiles each)
// ---------------------------------------------------------------------------
__global__ __launch_bounds__(256) void def_gemm(
    const float* __restrict__ attn2, const float* __restrict__ offs,
    const ushort_t* __restrict__ afrag, const float* __restrict__ bias,
    float* __restrict__ out)
{
    __shared__ __align__(32) ushort_t bfrag[2][2][512];
    __shared__ int   cidx[27*32*8];
    __shared__ float cwt [27*32*8];
    const int col0 = blockIdx.x * 32, tid = threadIdx.x;
    const int lane = tid & 31, wv = tid >> 5;

    // Per-(tap, voxel) 8-corner table (channel independent), once per block
    for (int e = tid; e < 27*32; e += 256) {
        int tap = e >> 5, nn = e & 31;
        int n = col0 + nn;
        int z = n / PLANE, y = (n % PLANE) / DIMS, x = n % DIMS;
        float cz = (float)z + (float)(tap/9     - 1) + offs[(tap*3 + 0)*V3 + n];
        float cy = (float)y + (float)((tap/3)%3 - 1) + offs[(tap*3 + 1)*V3 + n];
        float cx = (float)x + (float)(tap%3     - 1) + offs[(tap*3 + 2)*V3 + n];
        float fz = floorf(cz), fy = floorf(cy), fx = floorf(cx);
        int c0z = (int)fz, c0y = (int)fy, c0x = (int)fx;
        float rz = cz - fz, ry = cy - fy, rx = cx - fx;
        for (int j = 0; j < 8; ++j) {
            int dz = j >> 2, dy = (j >> 1) & 1, dx = j & 1;
            int iz = c0z + dz, iy = c0y + dy, ix = c0x + dx;
            bool valid = (unsigned)iz < DIMS && (unsigned)iy < DIMS && (unsigned)ix < DIMS;
            float w = (dz ? rz : 1.0f - rz) * (dy ? ry : 1.0f - ry) * (dx ? rx : 1.0f - rx);
            cidx[e*8 + j] = valid ? (iz*DIMS + iy)*DIMS + ix : 0;
            cwt [e*8 + j] = valid ? w : 0.0f;
        }
    }

    auto build = [&](int p, int kt) {
        for (int e2 = tid; e2 < 2*512; e2 += 256) {
            int st = e2 >> 9, e = e2 & 511;
            int nn, kk; b_elem(e, nn, kk);
            int k = kt*32 + kk;
            int c = k / 27, tap = k % 27;
            const float* ac = attn2 + c * V3;
            int base = (tap*32 + st*16 + nn) * 8;
            float v = 0.0f;
            #pragma unroll
            for (int j = 0; j < 8; ++j) v += cwt[base + j] * ac[cidx[base + j]];
            bfrag[p][st][e] = f2bfu(v);
        }
    };

    __syncthreads();           // corner table ready
    v8f acc0 = {}, acc1 = {};
    build(0, 0);
    __syncthreads();
    for (int kt = 0; kt < KT_BIG; ++kt) {
        int cur = kt & 1;
        {
            v16bf a = *(const v16bf*)(afrag + ((size_t)(wv*KT_BIG + kt)*512 + lane*16));
            v16bf b0 = *(const v16bf*)(&bfrag[cur][0][lane*16]);
            v16bf b1 = *(const v16bf*)(&bfrag[cur][1][lane*16]);
            acc0 = __builtin_amdgcn_wmma_f32_16x16x32_bf16(false, a, false, b0, (short)0, acc0, false, false);
            acc1 = __builtin_amdgcn_wmma_f32_16x16x32_bf16(false, a, false, b1, (short)0, acc1, false, false);
        }
        if (kt + 1 < KT_BIG) build(cur ^ 1, kt + 1);
        __syncthreads();
    }
    int mhi = (lane >> 4) << 3;
    v8f accs[2] = {acc0, acc1};
    for (int st = 0; st < 2; ++st) {
        int n = col0 + st*16 + (lane & 15);
        for (int r = 0; r < 8; ++r) {
            int m = wv*16 + r + mhi;
            out[m*V3 + n] = accs[st][r] + bias[m];
        }
    }
}

// ---------------------------------------------------------------------------
// 1x1x1 conv GEMM [128 x 128] x [128 x 64-col tile] + bias, fused x*attn
// grid = 216 x 256
// ---------------------------------------------------------------------------
__global__ __launch_bounds__(256) void fin_gemm(
    const float* __restrict__ defout, const ushort_t* __restrict__ afrag,
    const float* __restrict__ bias, const float* __restrict__ xin,
    float* __restrict__ out)
{
    __shared__ __align__(32) ushort_t bfrag[2][4][512];
    const int col0 = blockIdx.x * 64, tid = threadIdx.x;
    const int lane = tid & 31, wv = tid >> 5;

    auto build = [&](int p, int kt) {
        for (int e4 = tid; e4 < 4*512; e4 += 256) {
            int st = e4 >> 9, e = e4 & 511;
            int nn, kk; b_elem(e, nn, kk);
            int k = kt*32 + kk;
            bfrag[p][st][e] = f2bfu(defout[k*V3 + col0 + st*16 + nn]);
        }
    };

    v8f acc0 = {}, acc1 = {}, acc2 = {}, acc3 = {};
    build(0, 0);
    __syncthreads();
    for (int kt = 0; kt < KT_ONE; ++kt) {
        int cur = kt & 1;
        {
            v16bf a = *(const v16bf*)(afrag + ((size_t)(wv*KT_ONE + kt)*512 + lane*16));
            v16bf b0 = *(const v16bf*)(&bfrag[cur][0][lane*16]);
            v16bf b1 = *(const v16bf*)(&bfrag[cur][1][lane*16]);
            v16bf b2 = *(const v16bf*)(&bfrag[cur][2][lane*16]);
            v16bf b3 = *(const v16bf*)(&bfrag[cur][3][lane*16]);
            acc0 = __builtin_amdgcn_wmma_f32_16x16x32_bf16(false, a, false, b0, (short)0, acc0, false, false);
            acc1 = __builtin_amdgcn_wmma_f32_16x16x32_bf16(false, a, false, b1, (short)0, acc1, false, false);
            acc2 = __builtin_amdgcn_wmma_f32_16x16x32_bf16(false, a, false, b2, (short)0, acc2, false, false);
            acc3 = __builtin_amdgcn_wmma_f32_16x16x32_bf16(false, a, false, b3, (short)0, acc3, false, false);
        }
        if (kt + 1 < KT_ONE) build(cur ^ 1, kt + 1);
        __syncthreads();
    }
    int mhi = (lane >> 4) << 3;
    v8f accs[4] = {acc0, acc1, acc2, acc3};
    for (int st = 0; st < 4; ++st) {
        int n = col0 + st*16 + (lane & 15);
        for (int r = 0; r < 8; ++r) {
            int m = wv*16 + r + mhi;
            int i = m*V3 + n;
            out[i] = xin[i] * (accs[st][r] + bias[m]);
        }
    }
}

// ---------------------------------------------------------------------------
extern "C" void kernel_launch(void* const* d_in, const int* in_sizes, int n_in,
                              void* d_out, int out_size, void* d_ws, size_t ws_size,
                              hipStream_t stream) {
    const float* x     = (const float*)d_in[0];
    const float* w0    = (const float*)d_in[1];
    const float* b0    = (const float*)d_in[2];
    const float* wsd   = (const float*)d_in[3];   // dilated depthwise weights
    const float* bs    = (const float*)d_in[4];
    const float* w_off = (const float*)d_in[5];
    const float* b_off = (const float*)d_in[6];
    const float* w_def = (const float*)d_in[7];
    const float* b_def = (const float*)d_in[8];
    const float* w1    = (const float*)d_in[9];
    const float* b1    = (const float*)d_in[10];
    float* out = (float*)d_out;

    // Workspace layout (bytes, all 256-aligned)
    char* ws = (char*)d_ws;
    const size_t SZ_MAP  = (size_t)128 * V3 * 4;          // 7,077,888
    const size_t SZ_OFF  = (size_t)81  * V3 * 4;          // 4,478,976
    float*    attn1  = (float*)(ws);                      // reused as defout later
    float*    attn2  = (float*)(ws + SZ_MAP);
    float*    offs   = (float*)(ws + 2*SZ_MAP);
    ushort_t* fdef   = (ushort_t*)(ws + 2*SZ_MAP + SZ_OFF);
    ushort_t* foff   = (ushort_t*)((char*)fdef + (size_t)N_FDEF * 2);
    ushort_t* f1     = (ushort_t*)((char*)foff + (size_t)N_FOFF * 2);
    float*    defout = attn1;   // attn1 dead after attn2 + offsets computed

    // 1) pre-swizzle GEMM weights to bf16 A-fragments
    {
        int total = N_FDEF + N_FOFF + N_F1;
        prep_frags<<<(total + 255) / 256, 256, 0, stream>>>(w_def, w_off, w1,
                                                            fdef, foff, f1);
    }
    // 2) depthwise 5^3 (dil=1, pad=2)
    dw5<<<dim3(V3/256, 128), 256, 0, stream>>>(x, w0, b0, attn1, 1);
    // 3) depthwise dilated 5^3 (dil=3, pad=6)
    dw5<<<dim3(V3/256, 128), 256, 0, stream>>>(attn1, wsd, bs, attn2, 3);
    // 4) offset conv (WMMA implicit GEMM, 64 cols/block)
    off_gemm<<<V3/64, 256, 0, stream>>>(attn2, foff, b_off, offs);
    // 5) deformable conv (trilinear sample + WMMA GEMM, 32 cols/block)
    def_gemm<<<V3/32, 256, 0, stream>>>(attn2, offs, fdef, b_def, defout);
    // 6) 1x1x1 conv + bias + x*attn fused (64 cols/block)
    fin_gemm<<<V3/64, 256, 0, stream>>>(defout, f1, b1, x, out);
}
